// GlobalLocalModel_56899726737982
// MI455X (gfx1250) — matrix operand order repaired and checked by reading, not compiled, not run on hardware
//
#include <hip/hip_runtime.h>
#include <hip/hip_bf16.h>
#include <math.h>

// ---------------------------------------------------------------------------
// Types
// ---------------------------------------------------------------------------
typedef __attribute__((ext_vector_type(16))) _Float16 v16h;
typedef __attribute__((ext_vector_type(8)))  _Float16 v8h;
typedef __attribute__((ext_vector_type(8)))  float    v8f;
typedef __attribute__((ext_vector_type(4)))  unsigned int u32x4;
typedef __attribute__((ext_vector_type(8)))  int i32x8;
typedef __attribute__((ext_vector_type(4)))  int i32x4;

#define BATCH     8
#define IMG       1024
#define CH        3
#define PS        64
#define KSEL      256
#define GRIDW     121            // (1024-64)/8 + 1
#define NSC       (GRIDW*GRIDW)  // 14641
#define CW        256            // coarse width (1024/4), padded GEMM dim
#define VALIDW    241            // 256 - 16 + 1

// ---------------------------------------------------------------------------
// Stage 1: 4x4 average pool of labels -> f16 coarse map, stored TRANSPOSED
// (coarseT[x][y]) so the GEMM B-fragment loads are contiguous 16B loads (two per lane).
// ---------------------------------------------------------------------------
__global__ void pool4_kernel(const float* __restrict__ labels,
                             _Float16* __restrict__ coarseT) {
  int idx = blockIdx.x * blockDim.x + threadIdx.x;     // 8*256*256 threads
  int b = idx >> 16;
  int r = (idx >> 8) & 255;
  int c = idx & 255;
  const float* L = labels + ((long)b << 20) + (long)(r * 4) * IMG + c * 4;
  float s = 0.0f;
#pragma unroll
  for (int dy = 0; dy < 4; ++dy)
#pragma unroll
    for (int dx = 0; dx < 4; ++dx) s += L[dy * IMG + dx];
  coarseT[((long)b << 16) + c * CW + r] = (_Float16)(s * 0.0625f);
}

// ---------------------------------------------------------------------------
// Stage 2: banded ones matrix for the separable 16x16 sliding box-sum.
//   S[i][j] = 1 if i<241 && i<=j<i+16
// Note: transpose(S^T) == S, so S doubles as the BT operand of GEMM2.
// ---------------------------------------------------------------------------
__global__ void bands_kernel(_Float16* __restrict__ Smat) {
  int idx = blockIdx.x * blockDim.x + threadIdx.x;     // 65536 threads
  int i = idx >> 8, j = idx & 255;
  Smat[idx] = (_Float16)((i < VALIDW && j >= i && j < i + 16) ? 1.0f : 0.0f);
}

// ---------------------------------------------------------------------------
// Stage 3: 256x256x256 GEMM via V_WMMA_F32_16X16X32_F16.
// One wave computes one 16x16 C tile. B is supplied TRANSPOSED (BT, N x K
// row-major) so every fragment load is a contiguous 16/32-byte load:
//   A  halves 0-7 : A[m][k0+hi*8 .. +7]      (b128)
//   A  halves 8-15: A[m][k0+16+hi*8 .. +7]   (b128)
//   BT halves 0-7 : BT[n][k0+hi*8 .. +7]     (b128)
//   BT halves 8-15: BT[n][k0+16+hi*8 .. +7]  (b128)
// k-loop fully unrolled -> 8 static v_wmma per kernel.
// ---------------------------------------------------------------------------
__global__ __launch_bounds__(32) void wmma_gemm_256(
    const _Float16* __restrict__ A, long aBatchStride,
    const _Float16* __restrict__ BT, long bBatchStride,
    void* __restrict__ Out, long oBatchStride, int outHalf) {
  int tile = blockIdx.x & 255;
  int b    = blockIdx.x >> 8;
  int m0 = (tile >> 4) << 4;
  int n0 = (tile & 15) << 4;
  int lane = threadIdx.x & 31;
  int hi   = (lane >= 16) ? 1 : 0;
  int mrow = m0 + (lane & 15);
  int ncol = n0 + (lane & 15);
  const _Float16* Arow = A + (long)b * aBatchStride + (long)mrow * CW;
  const _Float16* Brow = BT + (long)b * bBatchStride + (long)ncol * CW;

  v8f acc = {};
#pragma unroll
  for (int k0 = 0; k0 < CW; k0 += 32) {
    v8h a0 = *(const v8h*)(Arow + k0 + hi * 8);
    v8h a1 = *(const v8h*)(Arow + k0 + 16 + hi * 8);
    v16h afrag = __builtin_shufflevector(a0, a1, 0, 1, 2, 3, 4, 5, 6, 7,
                                         8, 9, 10, 11, 12, 13, 14, 15);
    v8h b0 = *(const v8h*)(Brow + k0 + hi * 8);
    v8h b1 = *(const v8h*)(Brow + k0 + 16 + hi * 8);
    v16h bfrag = __builtin_shufflevector(b0, b1, 0, 1, 2, 3, 4, 5, 6, 7,
                                         8, 9, 10, 11, 12, 13, 14, 15);
    acc = __builtin_amdgcn_wmma_f32_16x16x32_f16(
        false, afrag, false, bfrag, (short)0, acc, false, false);
  }

  if (outHalf) {
    _Float16* O = (_Float16*)Out + (long)b * oBatchStride;
#pragma unroll
    for (int v = 0; v < 8; ++v) {
      int M = m0 + v + (hi ? 8 : 0);
      O[(long)M * CW + ncol] = (_Float16)acc[v];
    }
  } else {
    float* O = (float*)Out + (long)b * oBatchStride;
#pragma unroll
    for (int v = 0; v < 8; ++v) {
      int M = m0 + v + (hi ? 8 : 0);
      O[(long)M * CW + ncol] = acc[v];
    }
  }
}

// ---------------------------------------------------------------------------
// Stage 4: per-batch: bilinear 241->121 resize, min/max normalize, /TEMP,
// +Gumbel, then 256 deterministic argmax sweeps (top-k). One block per batch.
// Scores live in LDS (58.8 KB -- comfortable within the 320 KB WGP LDS).
// ---------------------------------------------------------------------------
__global__ __launch_bounds__(1024) void select_topk(
    const float* __restrict__ scores, int* __restrict__ selhw,
    float* __restrict__ coords) {
  __shared__ float sc[NSC];
  __shared__ float rmax[32];
  __shared__ int   rimax[32];
  __shared__ float bstat[2];
  int b = blockIdx.x, t = threadIdx.x;
  int wave = t >> 5, lane = t & 31;
  const float* S = scores + (long)b * (CW * CW);

  const float scale = 241.0f / 121.0f;
  for (int i = t; i < NSC; i += 1024) {
    int oy = i / GRIDW, ox = i - oy * GRIDW;
    float cy = (oy + 0.5f) * scale - 0.5f;
    cy = fminf(fmaxf(cy, 0.0f), 240.0f);
    int y0 = (int)floorf(cy); int y1 = min(y0 + 1, 240); float fy = cy - y0;
    float cx = (ox + 0.5f) * scale - 0.5f;
    cx = fminf(fmaxf(cx, 0.0f), 240.0f);
    int x0 = (int)floorf(cx); int x1 = min(x0 + 1, 240); float fx = cx - x0;
    float v00 = S[y0 * CW + x0], v01 = S[y0 * CW + x1];
    float v10 = S[y1 * CW + x0], v11 = S[y1 * CW + x1];
    float top = v00 * (1.0f - fx) + v01 * fx;
    float bot = v10 * (1.0f - fx) + v11 * fx;
    sc[i] = top * (1.0f - fy) + bot * fy;
  }
  __syncthreads();

  // block min/max (wave32 shuffles)
  float mn = 3.4e38f, mx = -3.4e38f;
  for (int i = t; i < NSC; i += 1024) {
    float v = sc[i];
    mn = fminf(mn, v);
    mx = fmaxf(mx, v);
  }
  for (int off = 16; off > 0; off >>= 1) {
    mn = fminf(mn, __shfl_down(mn, off, 32));
    mx = fmaxf(mx, __shfl_down(mx, off, 32));
  }
  if (lane == 0) { rmax[wave] = mx; rimax[wave] = __float_as_int(mn); }
  __syncthreads();
  if (wave == 0) {
    float m2 = rmax[lane];
    float n2 = __int_as_float(rimax[lane]);
    for (int off = 16; off > 0; off >>= 1) {
      m2 = fmaxf(m2, __shfl_down(m2, off, 32));
      n2 = fminf(n2, __shfl_down(n2, off, 32));
    }
    if (lane == 0) { bstat[0] = n2; bstat[1] = m2; }
  }
  __syncthreads();
  float smin = bstat[0];
  float rng  = fmaxf(bstat[1] - smin, 1e-6f);
  float inv  = 1.0f / (rng * 0.1f);   // includes /TEMP

  // deterministic hash-based Gumbel noise (JAX threefry not reproducible here)
  for (int i = t; i < NSC; i += 1024) {
    unsigned u = (unsigned)(b * NSC + i) * 2654435761u;
    u ^= 61u ^ (u >> 16); u *= 9u; u ^= u >> 4; u *= 0x27d4eb2du; u ^= u >> 15;
    float un = ((float)(u >> 8) + 0.5f) * (1.0f / 16777216.0f);
    float g  = -logf(-logf(un));
    sc[i] = (sc[i] - smin) * inv + g;
  }
  __syncthreads();

  for (int k = 0; k < KSEL; ++k) {
    float bv = -3.4e38f; int bi = 0;
    for (int i = t; i < NSC; i += 1024) {
      float v = sc[i];
      if (v > bv) { bv = v; bi = i; }
    }
    for (int off = 16; off > 0; off >>= 1) {
      float ov = __shfl_down(bv, off, 32);
      int   oi = __shfl_down(bi, off, 32);
      if (ov > bv) { bv = ov; bi = oi; }
    }
    if (lane == 0) { rmax[wave] = bv; rimax[wave] = bi; }
    __syncthreads();
    if (t == 0) {
      float gb = rmax[0]; int gi = rimax[0];
      for (int wv = 1; wv < 32; ++wv)
        if (rmax[wv] > gb) { gb = rmax[wv]; gi = rimax[wv]; }
      sc[gi] = -3.4e38f;
      int row = gi / GRIDW, col = gi - row * GRIDW;
      int hs = row * 8, wsv = col * 8;
      selhw[(b * KSEL + k) * 2 + 0] = hs;
      selhw[(b * KSEL + k) * 2 + 1] = wsv;
      coords[(b * KSEL + k) * 2 + 0] = (float)hs;
      coords[(b * KSEL + k) * 2 + 1] = (float)wsv;
    }
    __syncthreads();
  }
}

// ---------------------------------------------------------------------------
// Stage 5: patch gather via Tensor Data Mover.
// One wave per (batch, patch, plane): TDM pulls the strided 64x64 f32 tile
// into LDS, then TDM flushes 16 KB contiguously to the output.
// Descriptor bitfields per CDNA5 ISA.
// ---------------------------------------------------------------------------
__device__ __forceinline__ void tdm_load(u32x4 g0, i32x8 g1, i32x4 g2, i32x4 g3) {
#if defined(__clang_major__) && (__clang_major__ >= 23)
  i32x8 g4 = {};
  __builtin_amdgcn_tensor_load_to_lds(g0, g1, g2, g3, g4, 0);
#else
  __builtin_amdgcn_tensor_load_to_lds(g0, g1, g2, g3, 0);
#endif
}

__device__ __forceinline__ void tdm_store(u32x4 g0, i32x8 g1, i32x4 g2, i32x4 g3) {
#if defined(__clang_major__) && (__clang_major__ >= 23)
  i32x8 g4 = {};
  __builtin_amdgcn_tensor_store_from_lds(g0, g1, g2, g3, g4, 0);
#else
  __builtin_amdgcn_tensor_store_from_lds(g0, g1, g2, g3, 0);
#endif
}

__global__ __launch_bounds__(32) void gather_tdm(
    const float* __restrict__ image, const float* __restrict__ labels,
    const int* __restrict__ selhw, float* __restrict__ outP,
    float* __restrict__ outL) {
  __shared__ float tilebuf[PS * PS];
  // generic LDS pointer: low 32 bits are the LDS byte offset (ISA §10.2)
  unsigned lds_off = (unsigned)(uintptr_t)(void*)tilebuf;

  int blk = blockIdx.x;                 // B * 256 * 4 blocks
  int b   = blk >> 10;
  int rem = blk & 1023;
  int k   = rem >> 2;
  int p   = rem & 3;

  int h = __builtin_amdgcn_readfirstlane(selhw[(b * KSEL + k) * 2 + 0]);
  int w = __builtin_amdgcn_readfirstlane(selhw[(b * KSEL + k) * 2 + 1]);

  const float* src;
  float* dst;
  if (p < 3) {
    src = image + (((long)(b * CH + p)) << 20) + (long)h * IMG + w;
    dst = outP + ((long)(b * KSEL + k) * CH + p) * (PS * PS);
  } else {
    src = labels + ((long)b << 20) + (long)h * IMG + w;
    dst = outL + (long)(b * KSEL + k) * (PS * PS);
  }

  // ---- load descriptor: 64x64 f32 tile, row stride 1024 elements ----
  unsigned long long ga = (unsigned long long)(uintptr_t)src;
  u32x4 g0 = { 1u,                                   // count=1, user mode
               lds_off,                              // lds_addr
               (unsigned)(ga & 0xFFFFFFFFu),         // global_addr[31:0]
               (unsigned)((ga >> 32) & 0x1FFFFFFu) | (2u << 30) }; // addr[56:32] | type=2
  i32x8 g1 = { 0x20000,                // data_size=4B, mask=0
               0,                      // atomic barrier addr
               1,                      // tensor_dim0 = 65536 -> [31:16]=1 here
               1 | (64 << 16),         // tensor_dim1 hi16=1 (65536), tile_dim0=64
               64,                     // tile_dim1=64, tile_dim2=0
               1024,                   // tensor_dim0_stride[31:0] = 1024
               0, 0 };                 // stride hi bits, tensor_dim1_stride
  i32x4 gz = {};
  tdm_load(g0, g1, gz, gz);
  __builtin_amdgcn_s_wait_tensorcnt(0);

  // ---- store descriptor: contiguous 4096-element 1D tile ----
  unsigned long long gb = (unsigned long long)(uintptr_t)dst;
  u32x4 s0 = { 1u, lds_off,
               (unsigned)(gb & 0xFFFFFFFFu),
               (unsigned)((gb >> 32) & 0x1FFFFFFu) | (2u << 30) };
  i32x8 s1 = { 0x20000,
               0,
               1,                              // tensor_dim0 = 65536
               1 | (int)(4096u << 16),         // tile_dim0 = 4096
               0,                              // tile_dim1 unused (1D)
               4096,
               0, 0 };
  tdm_store(s0, s1, gz, gz);
  __builtin_amdgcn_s_wait_tensorcnt(0);
}

// ---------------------------------------------------------------------------
// Host launch
// ---------------------------------------------------------------------------
extern "C" void kernel_launch(void* const* d_in, const int* in_sizes, int n_in,
                              void* d_out, int out_size, void* d_ws, size_t ws_size,
                              hipStream_t stream) {
  const float* image  = (const float*)d_in[0];   // (8,3,1024,1024)
  const float* labels = (const float*)d_in[1];   // (8,1,1024,1024)

  char* wp = (char*)d_ws;
  auto carve = [&](size_t bytes) {
    void* p = (void*)wp;
    wp += (bytes + 255) & ~(size_t)255;
    return p;
  };
  _Float16* coarseT = (_Float16*)carve((size_t)BATCH * CW * CW * 2);
  _Float16* Smat    = (_Float16*)carve((size_t)CW * CW * 2);
  _Float16* T1      = (_Float16*)carve((size_t)BATCH * CW * CW * 2);
  float*    scores  = (float*)carve((size_t)BATCH * CW * CW * 4);
  int*      selhw   = (int*)carve((size_t)BATCH * KSEL * 2 * 4);

  float* out  = (float*)d_out;
  float* outP = out;                                        // 8*256*3*64*64
  float* outL = out + (size_t)BATCH * KSEL * CH * PS * PS;  // 8*256*1*64*64
  float* outC = outL + (size_t)BATCH * KSEL * PS * PS;      // 8*256*2

  pool4_kernel<<<(BATCH * CW * CW) / 256, 256, 0, stream>>>(labels, coarseT);
  bands_kernel<<<(CW * CW) / 256, 256, 0, stream>>>(Smat);
  // T1 = S * coarse     (B supplied transposed = coarseT; f16 out for chaining)
  wmma_gemm_256<<<BATCH * 256, 32, 0, stream>>>(Smat, 0, coarseT, CW * CW,
                                                (void*)T1, CW * CW, 1);
  // scores = T1 * S^T   (BT of S^T is S itself; f32 out)
  wmma_gemm_256<<<BATCH * 256, 32, 0, stream>>>(T1, CW * CW, Smat, 0,
                                                (void*)scores, CW * CW, 0);
  select_topk<<<BATCH, 1024, 0, stream>>>(scores, selhw, outC);
  gather_tdm<<<BATCH * KSEL * 4, 32, 0, stream>>>(image, labels, selhw, outP, outL);
}
